// IMG_2095944041153
// MI455X (gfx1250) — matrix operand order repaired and checked by reading, not compiled, not run on hardware
//
#include <hip/hip_runtime.h>
#include <hip/hip_bf16.h>

// Problem dims (fixed by reference)
#define B_   64
#define T_   1024
#define D_   256
#define H_   512
#define NWG  16       // one WG per 32 output columns; H_/32 == 16
#define NTHR 256      // 8 wave32 waves per WG

typedef __attribute__((ext_vector_type(16))) _Float16 v16h;
typedef __attribute__((ext_vector_type(8)))  float    v8f;
typedef __attribute__((ext_vector_type(4)))  int      v4i;

union V16H { v16h v; unsigned u[8]; _Float16 h[16]; };

// CDNA5 async global->LDS copy (ASYNCcnt-tracked), with fallback.
// Toolchain signature (from diagnostics): (v4i AS1*, v4i AS3*, imm, imm)
#if defined(__has_builtin)
#if __has_builtin(__builtin_amdgcn_global_load_async_to_lds_b128) && \
    __has_builtin(__builtin_amdgcn_s_wait_asynccnt)
#define USE_ASYNC_LDS 1
#endif
#endif

#ifdef USE_ASYNC_LDS
typedef __attribute__((address_space(1))) v4i GV4;
typedef __attribute__((address_space(3))) v4i LV4;
__device__ __forceinline__ void async_copy16(const void* g, void* l) {
  __builtin_amdgcn_global_load_async_to_lds_b128(
      (GV4*)g, (LV4*)l, /*imm_offset=*/0, /*cpol=*/0);
}
#endif

// ---------------- LDS layout (units: _Float16 elements) -------------------
// Padded strides chosen so row-parallel accesses rotate LDS banks:
//   520*2B = 1040B = 260 dwords; 260 % 64 = 4  -> 4-bank rotation per row
//   264*2B =  528B = 132 dwords; 132 % 64 = 4  -> 4-bank rotation per row
#define XS 264   // x tile      [64][256+8]
#define SS 520   // state       [64][512+8]
#define W5 520   // K=512 weights, transposed [32 cols][512+8]
#define W2 264   // K=256 weights, transposed [32 cols][256+8]

#define OFF_X   0
#define OFF_S   (OFF_X  + 64*XS)
#define OFF_P   (OFF_S  + 64*SS)
#define OFF_UP  (OFF_P  + 64*SS)
#define OFF_UH  (OFF_UP + 32*W5)
#define OFF_GP  (OFF_UH + 32*W5)
#define OFF_WP  (OFF_GP + 32*W5)
#define OFF_WH  (OFF_WP + 32*W2)
#define LDS_F16 (OFF_WH + 32*W2)   // 150272 f16 = 300544 B  (<= 320KB WGP LDS)

// ---------------- WMMA helpers (CDNA5 VGPR layouts, wave32) ---------------
__device__ __forceinline__ v8f wmma_f16(v16h a, v16h b, v8f c) {
  // D = A(16x32 f16) * B(32x16 f16) + C(16x16 f32)
  return __builtin_amdgcn_wmma_f32_16x16x32_f16(
      /*neg_a=*/false, a, /*neg_b=*/false, b,
      /*c_mod=*/(short)0, c, /*reuse_a=*/false, /*reuse_b=*/false);
}

// A-matrix 16x32 f16 layout:
//  lane half h (0: lanes 0-15, 1: lanes 16-31), row M = lane&15
//  VGPR g in 0..3: K = 2g + 8h (+0,+1) ; g in 4..7: K = 16 + 2(g-4) + 8h
__device__ __forceinline__ v16h lds_load_A(const _Float16* base, int stride,
                                           int m0, int kbase, int lane) {
  const int row  = m0 + (lane & 15);
  const int half = lane >> 4;
  const unsigned* p = (const unsigned*)(base + row * stride + kbase);
  V16H r;
#pragma unroll
  for (int g = 0; g < 8; ++g) {
    const int k0 = ((g < 4) ? (g << 1) : (16 + ((g - 4) << 1))) + (half << 3);
    r.u[g] = p[k0 >> 1];
  }
  return r.v;
}

// B-matrix 32x16 f16, with the weight slice stored TRANSPOSED in LDS
// ([32 cols][K]): per lane col N = lane&15 (+16-lane half selects K block).
//  VGPR g in 0..3: K = 2g + 8h (+0,+1) ; g in 4..7: K = 16 + 2(g-4) + 8h -> consecutive pair along K = one u32 load.
__device__ __forceinline__ v16h lds_load_Bt(const _Float16* base, int stride,
                                            int n0, int kbase, int lane) {
  const int col  = n0 + (lane & 15);
  const int half = lane >> 4;
  const unsigned* p = (const unsigned*)(base + col * stride + kbase);
  V16H r;
#pragma unroll
  for (int g = 0; g < 8; ++g) {
    const int k0 = ((g < 4) ? (g << 1) : (16 + ((g - 4) << 1))) + (half << 3);
    r.u[g] = p[k0 >> 1];
  }
  return r.v;
}

// ---------------- cooperative loaders -------------------------------------
__device__ __forceinline__ void load_x_tile(const float* __restrict__ x, int t,
                                            _Float16* __restrict__ xs, int tid) {
  // x[b][t][d], load the [64][256] f32 slice for timestep t, convert to f16
#pragma unroll 4
  for (int i = tid; i < (B_ * D_) / 4; i += NTHR) {
    const int e = i << 2;
    const int row = e >> 8, d = e & (D_ - 1);
    const float4 v = *(const float4*)(x + ((size_t)row * T_ + t) * D_ + d);
    _Float16* o = xs + row * XS + d;
    o[0] = (_Float16)v.x; o[1] = (_Float16)v.y;
    o[2] = (_Float16)v.z; o[3] = (_Float16)v.w;
  }
}

__device__ __forceinline__ void load_state(const _Float16* __restrict__ sg,
                                           const _Float16* __restrict__ pg,
                                           _Float16* __restrict__ ss,
                                           _Float16* __restrict__ ps, int tid) {
  // state buffers are [64][512] f16 contiguous in global; copy into padded LDS
#pragma unroll 4
  for (int i = tid; i < (B_ * H_) / 8; i += NTHR) {
    const int e = i << 3;
    const int row = e >> 9, c = e & (H_ - 1);
#ifdef USE_ASYNC_LDS
    async_copy16(sg + e, ss + row * SS + c);   // global_load_async_to_lds_b128
    async_copy16(pg + e, ps + row * SS + c);
#else
    *(uint4*)(ss + row * SS + c) = *(const uint4*)(sg + e);
    *(uint4*)(ps + row * SS + c) = *(const uint4*)(pg + e);
#endif
  }
}

// ---------------- 16-WG device-wide barrier --------------------------------
__device__ __forceinline__ void grid_barrier(unsigned* __restrict__ sync) {
  __syncthreads();
  if (threadIdx.x == 0) {
    volatile unsigned* gen = sync + 1;
    __threadfence();
    const unsigned g = *gen;
    if (atomicAdd(sync, 1u) == NWG - 1u) {
      *sync = 0u;
      __threadfence();
      *gen = g + 1u;
    } else {
      while (*gen == g) { __builtin_amdgcn_s_sleep(1); }
    }
    __threadfence();
  }
  __syncthreads();
}

// ---------------- fused persistent RNN kernel ------------------------------
__global__ void __launch_bounds__(NTHR, 1)
rnn_scan_wmma(const float* __restrict__ x,  const float* __restrict__ wh,
              const float* __restrict__ uh, const float* __restrict__ bh,
              const float* __restrict__ wp, const float* __restrict__ up,
              const float* __restrict__ gp, const float* __restrict__ bp,
              const float* __restrict__ p0, const float* __restrict__ h0,
              float* __restrict__ out, unsigned* __restrict__ sync,
              _Float16* __restrict__ sg0, _Float16* __restrict__ sg1,
              _Float16* __restrict__ pg0, _Float16* __restrict__ pg1) {
  extern __shared__ _Float16 smem[];
  _Float16* xs  = smem + OFF_X;
  _Float16* ss  = smem + OFF_S;
  _Float16* ps  = smem + OFF_P;
  _Float16* ups = smem + OFF_UP;
  _Float16* uhs = smem + OFF_UH;
  _Float16* gps = smem + OFF_GP;
  _Float16* wps = smem + OFF_WP;
  _Float16* whs = smem + OFF_WH;

  const int tid  = threadIdx.x;
  const int wg   = blockIdx.x;         // owns columns [wg*32, wg*32+32)
  const int lane = tid & 31;
  const int wv   = tid >> 5;           // 8 waves

  // --- one-time: convert weight slices f32->f16, store transposed in LDS ---
  {
    const int c = tid & 31;            // column within slice (coalesced reads)
    const int k0 = tid >> 5;
    const int cg = wg * 32 + c;
    for (int k = k0; k < H_; k += 8) { // K=512 matrices
      ups[c * W5 + k] = (_Float16)up[(size_t)k * H_ + cg];
      uhs[c * W5 + k] = (_Float16)uh[(size_t)k * H_ + cg];
      gps[c * W5 + k] = (_Float16)gp[(size_t)k * H_ + cg];
    }
    for (int k = k0; k < D_; k += 8) { // K=256 matrices
      wps[c * W2 + k] = (_Float16)wp[(size_t)k * H_ + cg];
      whs[c * W2 + k] = (_Float16)wh[(size_t)k * H_ + cg];
    }
  }
  // --- one-time: initial state h0/p0 (f32) -> LDS f16 ---
  for (int e = tid * 4; e < B_ * H_; e += NTHR * 4) {
    const float4 hv = *(const float4*)(h0 + e);
    const float4 pv = *(const float4*)(p0 + e);
    const int row = e >> 9, c = e & (H_ - 1);
    _Float16* sd = ss + row * SS + c;
    sd[0] = (_Float16)hv.x; sd[1] = (_Float16)hv.y;
    sd[2] = (_Float16)hv.z; sd[3] = (_Float16)hv.w;
    _Float16* pd = ps + row * SS + c;
    pd[0] = (_Float16)pv.x; pd[1] = (_Float16)pv.y;
    pd[2] = (_Float16)pv.z; pd[3] = (_Float16)pv.w;
  }
  load_x_tile(x, 0, xs, tid);

  // --- per-wave tile assignment: 64x32 WG output = 4 (M) x 2 (N) tiles ---
  const int m0   = (wv & 3) << 4;          // batch-row base of 16x16 tile
  const int nloc = ((wv >> 2) & 1) << 4;   // column base within 32-col slice
  const int n_g  = wg * 32 + nloc + (lane & 15);
  const float biasP = bp[n_g];
  const float biasH = bh[n_g];
  __syncthreads();

  for (int t = 0; t < T_; ++t) {
    // Fused K-loops: each A tile is loaded once and feeds BOTH accumulators.
    //   accp = x_t@wp + s@up + pprev@gp   (-> sigmoid gate)
    //   acch = x_t@wh + s@uh              (-> tanh candidate)
    v8f accp = {};
    v8f acch = {};
#pragma unroll
    for (int kt = 0; kt < D_ / 32; ++kt) {
      v16h a   = lds_load_A(xs, XS, m0, kt * 32, lane);
      v16h bp_ = lds_load_Bt(wps, W2, nloc, kt * 32, lane);
      v16h bh_ = lds_load_Bt(whs, W2, nloc, kt * 32, lane);
      accp = wmma_f16(a, bp_, accp);
      acch = wmma_f16(a, bh_, acch);
    }
#pragma unroll
    for (int kt = 0; kt < H_ / 32; ++kt) {
      v16h a   = lds_load_A(ss, SS, m0, kt * 32, lane);
      v16h bu  = lds_load_Bt(ups, W5, nloc, kt * 32, lane);
      v16h bh_ = lds_load_Bt(uhs, W5, nloc, kt * 32, lane);
      accp = wmma_f16(a, bu, accp);
      acch = wmma_f16(a, bh_, acch);
      v16h a2 = lds_load_A(ps, SS, m0, kt * 32, lane);
      v16h bg = lds_load_Bt(gps, W5, nloc, kt * 32, lane);
      accp = wmma_f16(a2, bg, accp);
    }

    _Float16* snx = ((t + 1) & 1) ? sg1 : sg0;
    _Float16* pnx = ((t + 1) & 1) ? pg1 : pg0;
#pragma unroll
    for (int r = 0; r < 8; ++r) {
      // C/D layout: lane L holds (M = r + 8*(L>>4), N = L&15)
      const int m = m0 + r + ((lane >> 4) << 3);
      float pv = accp[r] + biasP;
      pv = 1.0f / (1.0f + __expf(-pv));
      const float hc = tanhf(acch[r] + biasH);
      const float sp = (float)ss[m * SS + n_g];
      const float hv = (1.0f - pv) * sp + pv * hc;
      out[((size_t)m * T_ + t) * H_ + n_g] = hv;        // [B,T,H] f32
      snx[m * H_ + n_g] = (_Float16)hv;                 // next-step state
      pnx[m * H_ + n_g] = (_Float16)pv;
    }
    __threadfence();
    grid_barrier(sync);          // all 16 WGs published the new state

    if (t + 1 < T_) {
      load_state(snx, pnx, ss, ps, tid);   // async b128 copies into LDS
      load_x_tile(x, t + 1, xs, tid);
      if (tid < B_) {            // warm L2 for x_{t+2}
        const int tp = (t + 2 < T_) ? (t + 2) : (T_ - 1);
        __builtin_prefetch(x + ((size_t)tid * T_ + tp) * D_, 0, 1);
      }
#ifdef USE_ASYNC_LDS
      __builtin_amdgcn_s_wait_asynccnt(0); // ASYNCcnt drained before reuse
#endif
    }
    __syncthreads();
  }
}

// ---------------- host entry ----------------------------------------------
extern "C" void kernel_launch(void* const* d_in, const int* in_sizes, int n_in,
                              void* d_out, int out_size, void* d_ws, size_t ws_size,
                              hipStream_t stream) {
  (void)in_sizes; (void)n_in; (void)out_size; (void)ws_size;
  const float* x  = (const float*)d_in[0];
  const float* wh = (const float*)d_in[1];
  const float* uh = (const float*)d_in[2];
  const float* bh = (const float*)d_in[3];
  const float* wp = (const float*)d_in[4];
  const float* up = (const float*)d_in[5];
  const float* gp = (const float*)d_in[6];
  const float* bp = (const float*)d_in[7];
  const float* p0 = (const float*)d_in[8];
  const float* h0 = (const float*)d_in[9];
  float* out = (float*)d_out;

  // workspace: [0,1024) sync, then 4 x 64KB f16 state buffers (double-buffered)
  unsigned char* ws = (unsigned char*)d_ws;
  unsigned* sync = (unsigned*)ws;
  const size_t SB = (size_t)B_ * H_ * sizeof(_Float16);  // 65536 B
  _Float16* sg0 = (_Float16*)(ws + 1024);
  _Float16* sg1 = (_Float16*)(ws + 1024 + SB);
  _Float16* pg0 = (_Float16*)(ws + 1024 + 2 * SB);
  _Float16* pg1 = (_Float16*)(ws + 1024 + 3 * SB);

  // zero barrier counter/generation every call (graph-replay deterministic)
  (void)hipMemsetAsync(d_ws, 0, 1024, stream);

  const size_t lds_bytes = (size_t)LDS_F16 * sizeof(_Float16);  // 300544 B
  rnn_scan_wmma<<<NWG, NTHR, lds_bytes, stream>>>(
      x, wh, uh, bh, wp, up, gp, bp, p0, h0, out, sync, sg0, sg1, pg0, pg1);
}
